// CausalAttention_14757507629137
// MI455X (gfx1250) — matrix operand order repaired and checked by reading, not compiled, not run on hardware
//
#include <hip/hip_runtime.h>

// ---------------------------------------------------------------------------
// CDNA5 (gfx1250) wave32 WMMA attention block.
// All matmuls run on v_wmma_f32_16x16x32_bf16 (bf16 in, fp32 accum).
// Fragment layouts used here (contiguous k, same order in A and B):
//   A (16x32 bf16): lane m=L%16, elem e     -> K = half*16+e
//   B (32x16 bf16): lane n=L%16, elem e     -> K = half*16+e
//   C/D (16x16 f32): VGPR r, lane L -> (M = r + 8*(L/16), N = L%16)
// LDS tile staging uses gfx1250 GLOBAL_LOAD_ASYNC_TO_LDS_B128 (ASYNCcnt) when
// the toolchain exposes the builtin, with a synchronous fallback.
// ---------------------------------------------------------------------------

typedef __bf16 v16bf __attribute__((ext_vector_type(16)));
typedef __bf16 v8bf  __attribute__((ext_vector_type(8)));
typedef float  v8f   __attribute__((ext_vector_type(8)));
typedef int    v4i   __attribute__((ext_vector_type(4)));

#define CAT16(lo, hi) __builtin_shufflevector((lo), (hi), 0,1,2,3,4,5,6,7,8,9,10,11,12,13,14,15)

static __device__ __forceinline__ v8f wmma_bf16(v16bf a, v16bf b, v8f c) {
  return __builtin_amdgcn_wmma_f32_16x16x32_bf16(false, a, false, b, (short)0, c, false, false);
}

// ---- gfx1250 async global->LDS copy (ASYNCcnt path) -----------------------
#ifndef __has_builtin
#define __has_builtin(x) 0
#endif

#if defined(__gfx1250__) && __has_builtin(__builtin_amdgcn_global_load_async_to_lds_b128)
#define USE_ASYNC_LDS 1
static __device__ __forceinline__ void cp16_async(void* lds, const void* g) {
  __builtin_amdgcn_global_load_async_to_lds_b128(
      (__attribute__((address_space(1))) v4i*)(v4i*)(void*)g,
      (__attribute__((address_space(3))) v4i*)(v4i*)lds, 0, 0);
}
static __device__ __forceinline__ void async_wait0() {
#if __has_builtin(__builtin_amdgcn_s_wait_asynccnt)
  __builtin_amdgcn_s_wait_asynccnt(0);
#else
  asm volatile("s_wait_asynccnt 0" ::: "memory");
#endif
}
#else
static __device__ __forceinline__ void cp16_async(void* lds, const void* g) {
  *(v8bf*)lds = *(const v8bf*)g;  // sync fallback: bounce through VGPRs
}
static __device__ __forceinline__ void async_wait0() {}
#endif

static __device__ __forceinline__ float redmax16(float v) {
  v = fmaxf(v, __shfl_xor(v, 1, 32));
  v = fmaxf(v, __shfl_xor(v, 2, 32));
  v = fmaxf(v, __shfl_xor(v, 4, 32));
  v = fmaxf(v, __shfl_xor(v, 8, 32));
  return v;
}
static __device__ __forceinline__ float redsum16(float v) {
  v += __shfl_xor(v, 1, 32);
  v += __shfl_xor(v, 2, 32);
  v += __shfl_xor(v, 4, 32);
  v += __shfl_xor(v, 8, 32);
  return v;
}

// ---------------------------------------------------------------------------
// Elementwise prep kernels
// ---------------------------------------------------------------------------
__global__ void cast_bf16_kernel(const float* __restrict__ in, __bf16* __restrict__ out, size_t n) {
  size_t i = (size_t)blockIdx.x * blockDim.x + threadIdx.x;
  if (i < n) out[i] = (__bf16)in[i];
}

// W[K][N] (row-major) -> Wt[N][K] bf16, so GEMM "B" tiles are K-contiguous.
__global__ void transpose_cast_kernel(const float* __restrict__ W, __bf16* __restrict__ Wt,
                                      int K, int N) {
  size_t i = (size_t)blockIdx.x * blockDim.x + threadIdx.x;
  if (i < (size_t)K * N) {
    int k = (int)(i / N), n = (int)(i % N);
    Wt[(size_t)n * K + k] = (__bf16)W[i];
  }
}

// Vt[b][h][d][t] = qkv[(b*T+t)*3C + 2C + h*64 + d]  (keys contiguous per (d) row)
__global__ void build_vt_kernel(const __bf16* __restrict__ qkv, __bf16* __restrict__ Vt) {
  constexpr int T = 2048, C3 = 3072, D = 64, H = 16;
  size_t i = (size_t)blockIdx.x * blockDim.x + threadIdx.x;  // over B*H*D*T
  int t = (int)(i % T); size_t rem = i / T;
  int d = (int)(rem % D); rem /= D;
  int h = (int)(rem % H);
  int b = (int)(rem / H);
  Vt[i] = qkv[((size_t)(b * T + t)) * C3 + 2048 + h * D + d];
}

// ---------------------------------------------------------------------------
// bf16 WMMA GEMM: C[M,N] = A[M,K] * Bt[N,K]^T.  MODE 0: bf16 out. MODE 1: f32 out + bias.
// Block 256 thr = 8 waves; tile 128x128x32; wave = 64x32 (4x2 WMMA tiles).
// Tiles staged via async global->LDS; next tile prefetched during compute.
// ---------------------------------------------------------------------------
template <int MODE>
__global__ void __launch_bounds__(256)
gemm_bf16_kernel(const __bf16* __restrict__ A, const __bf16* __restrict__ Bt,
                 void* __restrict__ Cp, const float* __restrict__ bias,
                 int M, int N, int K) {
  __shared__ alignas(64) __bf16 As[128][32];
  __shared__ alignas(64) __bf16 Bs[128][32];
  const int tid  = threadIdx.x;
  const int lane = tid & 31;
  const int ln = lane & 15, hf = lane >> 4;
  const int wv = tid >> 5;
  const int wm = wv >> 2, wn = wv & 3;
  const size_t m0 = (size_t)blockIdx.y * 128;
  const size_t n0 = (size_t)blockIdx.x * 128;

  // cooperative fill addressing: 512 x 16B chunks, 2 rows per thread
  const int r0 = tid >> 2, o0 = (tid & 3) << 3;     // rows r0 and r0+64
  const __bf16* gA0 = A  + (m0 + r0) * (size_t)K + o0;
  const __bf16* gA1 = A  + (m0 + r0 + 64) * (size_t)K + o0;
  const __bf16* gB0 = Bt + (n0 + r0) * (size_t)K + o0;
  const __bf16* gB1 = Bt + (n0 + r0 + 64) * (size_t)K + o0;

  v8f acc[4][2] = {};

  for (int k0 = 0; k0 < K; k0 += 32) {
    cp16_async(&As[r0][o0],      gA0);
    cp16_async(&As[r0 + 64][o0], gA1);
    cp16_async(&Bs[r0][o0],      gB0);
    cp16_async(&Bs[r0 + 64][o0], gB1);
    gA0 += 32; gA1 += 32; gB0 += 32; gB1 += 32;
    __builtin_prefetch(gA0, 0, 2);
    __builtin_prefetch(gA1, 0, 2);
    __builtin_prefetch(gB0, 0, 2);
    __builtin_prefetch(gB1, 0, 2);
    async_wait0();
    __syncthreads();

    v16bf bfr[2];
#pragma unroll
    for (int nt = 0; nt < 2; ++nt)
      bfr[nt] = *(const v16bf*)&Bs[wn * 32 + nt * 16 + ln][hf * 16];
#pragma unroll
    for (int mt = 0; mt < 4; ++mt) {
      const int ar = wm * 64 + mt * 16 + ln;
      v8bf lo = *(const v8bf*)&As[ar][hf * 16];
      v8bf hi = *(const v8bf*)&As[ar][hf * 16 + 8];
      v16bf af = CAT16(lo, hi);
#pragma unroll
      for (int nt = 0; nt < 2; ++nt)
        acc[mt][nt] = wmma_bf16(af, bfr[nt], acc[mt][nt]);
    }
    __syncthreads();
  }

#pragma unroll
  for (int mt = 0; mt < 4; ++mt)
#pragma unroll
    for (int nt = 0; nt < 2; ++nt)
#pragma unroll
      for (int r = 0; r < 8; ++r) {
        const size_t row = m0 + wm * 64 + mt * 16 + r + 8 * hf;
        const size_t col = n0 + wn * 32 + nt * 16 + ln;
        if constexpr (MODE == 0)
          ((__bf16*)Cp)[row * (size_t)N + col] = (__bf16)acc[mt][nt][r];
        else
          ((float*)Cp)[row * (size_t)N + col] = acc[mt][nt][r] + bias[col];
      }
}

// ---------------------------------------------------------------------------
// Flash attention, causal.  Block = 128 thr = 4 waves; wave owns 16 q rows.
// K/V tiles for each 32-key chunk staged ONCE per workgroup into LDS (async),
// shared by all 4 waves.  8 WMMAs per chunk (4 for S = Q K^T, 4 for P V).
// ---------------------------------------------------------------------------
__global__ void __launch_bounds__(128)
flash_attn_kernel(const __bf16* __restrict__ qkv, const __bf16* __restrict__ Vt,
                  __bf16* __restrict__ Oout) {
  constexpr int T = 2048, C3 = 3072, C = 1024, D = 64, H = 16;
  __shared__ alignas(64) __bf16 Ks[32][64];     // [key][d]
  __shared__ alignas(64) __bf16 Vs[64][32];     // [d][key]
  __shared__ alignas(64) __bf16 Pl[4][16][32];  // per-wave P transpose scratch

  const int tid  = threadIdx.x;
  const int lane = tid & 31;
  const int wv   = tid >> 5;
  const int ln = lane & 15, hf = lane >> 4;
  const int qb = blockIdx.x, h = blockIdx.y, b = blockIdx.z;
  const int q0 = qb * 64 + wv * 16;

  // cooperative staging: K tile 32x64 (rows kr, kr+16), V tile 64x32 (rows vr, vr+32)
  const int kr = tid >> 3, ko = (tid & 7) << 3;
  const int vr = tid >> 2, vo = (tid & 3) << 3;
  const __bf16* gK0 = qkv + ((size_t)(b * T + kr)) * C3 + C + h * D + ko;
  const __bf16* gK1 = gK0 + (size_t)16 * C3;
  const __bf16* gV0 = Vt + ((size_t)((b * H + h) * D + vr)) * T + vo;
  const __bf16* gV1 = gV0 + (size_t)32 * T;

  // Q fragments (A layout), contraction = d (two K=32 steps cover D=64)
  v16bf qf[2];
  {
    const __bf16* qp = qkv + ((size_t)(b * T + q0 + ln)) * C3 + h * D;
#pragma unroll
    for (int kk = 0; kk < 2; ++kk) {
      v8bf lo = *(const v8bf*)(qp + kk * 32 + hf * 16);
      v8bf hi = *(const v8bf*)(qp + kk * 32 + hf * 16 + 8);
      qf[kk] = CAT16(lo, hi);
    }
  }

  float Mr[8], Lr[8];
  v8f O[4] = {};
#pragma unroll
  for (int r = 0; r < 8; ++r) { Mr[r] = -1e30f; Lr[r] = 0.0f; }

  // v_exp_f32 is base-2: fold 1/sqrt(D) * log2(e) into the score scale
  const float scl2 = 0.125f * 1.44269504088896340736f;
  const int kend = qb * 64 + 64;

  for (int j0 = 0; j0 < kend; j0 += 32) {
    cp16_async(&Ks[kr][ko],      gK0);
    cp16_async(&Ks[kr + 16][ko], gK1);
    cp16_async(&Vs[vr][vo],      gV0);
    cp16_async(&Vs[vr + 32][vo], gV1);
    gK0 += (size_t)32 * C3; gK1 += (size_t)32 * C3; gV0 += 32; gV1 += 32;
    __builtin_prefetch(gK0, 0, 2);
    __builtin_prefetch(gK1, 0, 2);
    __builtin_prefetch(gV0, 0, 2);
    __builtin_prefetch(gV1, 0, 2);
    async_wait0();
    __syncthreads();

    // S tiles for keys [j0, j0+16) and [j0+16, j0+32); B-frags contiguous in LDS
    v8f S0 = {}, S1 = {};
#pragma unroll
    for (int kk = 0; kk < 2; ++kk) {
      S0 = wmma_bf16(qf[kk], *(const v16bf*)&Ks[ln][kk * 32 + hf * 16], S0);
      S1 = wmma_bf16(qf[kk], *(const v16bf*)&Ks[16 + ln][kk * 32 + hf * 16], S1);
    }

    // causal mask + online softmax (row m = r + 8*hf lives in one 16-lane half)
#pragma unroll
    for (int r = 0; r < 8; ++r) {
      const int row = q0 + r + 8 * hf;
      float s0 = (j0 + ln      <= row) ? S0[r] * scl2 : -1e30f;
      float s1 = (j0 + 16 + ln <= row) ? S1[r] * scl2 : -1e30f;
      float c  = redmax16(fmaxf(s0, s1));
      float Mn = fmaxf(Mr[r], c);
      float fac = __builtin_amdgcn_exp2f(Mr[r] - Mn);
      Mr[r] = Mn;
      float p0 = __builtin_amdgcn_exp2f(s0 - Mn);
      float p1 = __builtin_amdgcn_exp2f(s1 - Mn);
      Lr[r] = Lr[r] * fac + p0 + p1;
      Pl[wv][r + 8 * hf][ln]      = (__bf16)p0;
      Pl[wv][r + 8 * hf][16 + ln] = (__bf16)p1;
#pragma unroll
      for (int dt = 0; dt < 4; ++dt) O[dt][r] *= fac;
    }

    // P: D-layout -> A-layout via per-wave LDS (compiler inserts s_wait_dscnt)
    v8bf lo = *(const v8bf*)&Pl[wv][ln][hf * 16];
    v8bf hi = *(const v8bf*)&Pl[wv][ln][hf * 16 + 8];
    v16bf pf = CAT16(lo, hi);

#pragma unroll
    for (int dt = 0; dt < 4; ++dt)
      O[dt] = wmma_bf16(pf, *(const v16bf*)&Vs[dt * 16 + ln][hf * 16], O[dt]);

    __syncthreads();  // all waves done with Ks/Vs before next fill
  }

#pragma unroll
  for (int r = 0; r < 8; ++r) {
    const float inv = __builtin_amdgcn_rcpf(redsum16(Lr[r]));
    const size_t orow = ((size_t)(b * T + q0 + r + 8 * hf)) * C + h * D;
#pragma unroll
    for (int dt = 0; dt < 4; ++dt)
      Oout[orow + dt * 16 + ln] = (__bf16)(O[dt][r] * inv);
  }
}

// ---------------------------------------------------------------------------
// Host orchestration
// ---------------------------------------------------------------------------
extern "C" void kernel_launch(void* const* d_in, const int* in_sizes, int n_in,
                              void* d_out, int out_size, void* d_ws, size_t ws_size,
                              hipStream_t stream) {
  (void)in_sizes; (void)n_in; (void)out_size; (void)ws_size;
  const float* x     = (const float*)d_in[0];  // [4,2048,1024]
  const float* Wqkv  = (const float*)d_in[1];  // [1024,3072]
  const float* Wproj = (const float*)d_in[2];  // [1024,1024]
  const float* bproj = (const float*)d_in[3];  // [1024]
  float* out = (float*)d_out;                  // [4,2048,1024] f32

  constexpr int Bb = 4, T = 2048;
  constexpr size_t MT = (size_t)Bb * T;        // 8192 rows

  // workspace: xb 16MiB (reused as attn-out) | Wqkvt 6MiB | Wprojt 2MiB | qkv 48MiB | Vt 16MiB
  char* ws = (char*)d_ws;
  __bf16* xb     = (__bf16*)(ws);
  __bf16* Wqkvt  = (__bf16*)(ws + (16u << 20));
  __bf16* Wprojt = (__bf16*)(ws + (16u << 20) + (6u << 20));
  __bf16* qkvb   = (__bf16*)(ws + (16u << 20) + (6u << 20) + (2u << 20));
  __bf16* Vt     = (__bf16*)(ws + (16u << 20) + (6u << 20) + (2u << 20) + (48u << 20));

  cast_bf16_kernel<<<dim3((unsigned)((MT * 1024) / 256)), 256, 0, stream>>>(x, xb, MT * 1024);
  transpose_cast_kernel<<<dim3((1024u * 3072u) / 256u), 256, 0, stream>>>(Wqkv, Wqkvt, 1024, 3072);
  transpose_cast_kernel<<<dim3((1024u * 1024u) / 256u), 256, 0, stream>>>(Wproj, Wprojt, 1024, 1024);

  // qkv = x @ Wqkv   (bf16 out)
  gemm_bf16_kernel<0><<<dim3(3072 / 128, (unsigned)(MT / 128)), 256, 0, stream>>>(
      xb, Wqkvt, qkvb, nullptr, (int)MT, 3072, 1024);

  build_vt_kernel<<<dim3((unsigned)(((size_t)Bb * 16 * 64 * T) / 256)), 256, 0, stream>>>(qkvb, Vt);

  // attention output overwrites xb (GEMM1 already consumed it; stream-ordered)
  flash_attn_kernel<<<dim3(T / 64, 16, Bb), 128, 0, stream>>>(qkvb, Vt, xb);

  // out = attn_out @ Wproj + bproj  (f32 out)
  gemm_bf16_kernel<1><<<dim3(1024 / 128, (unsigned)(MT / 128)), 256, 0, stream>>>(
      xb, Wprojt, out, bproj, (int)MT, 1024, 1024);
}
